// GPUEfficientPathScoringModel_65481071410997
// MI455X (gfx1250) — 16-bit and 8-bit matrix operand order repaired and checked by reading, compiled, not run on hardware
//
#include <hip/hip_runtime.h>
#include <math.h>

#define HHD   4
#define CCH   64
#define DDM   256
#define HLU   64
#define LMAXC 256
#define LFMAXC 16

typedef __bf16 bf16_t;
typedef bf16_t v16bf __attribute__((ext_vector_type(16)));
typedef float  v8f   __attribute__((ext_vector_type(8)));
typedef int    v8i   __attribute__((ext_vector_type(8)));
typedef int    v2i   __attribute__((ext_vector_type(2)));
typedef int    v4i   __attribute__((ext_vector_type(4)));

__device__ __forceinline__ void atomicMaxF(float* addr, float val) {
  unsigned* ua = (unsigned*)addr;
  unsigned cur = *ua;
  while (__uint_as_float(cur) < val) {
    unsigned prev = cur;
    cur = atomicCAS(ua, prev, __float_as_uint(val));
    if (cur == prev) break;
  }
}

__global__ void k_fillf(float* p, float v, long n) {
  long i = (long)blockIdx.x * blockDim.x + threadIdx.x;
  if (i < n) p[i] = v;
}
__global__ void k_filli(int* p, int v, long n) {
  long i = (long)blockIdx.x * blockDim.x + threadIdx.x;
  if (i < n) p[i] = v;
}
__global__ void k_transpose(const float* __restrict__ W, float* __restrict__ WT, int K, int N) {
  long i = (long)blockIdx.x * blockDim.x + threadIdx.x;
  if (i >= (long)K * N) return;
  int k = (int)(i / N), n = (int)(i % N);
  WT[(long)n * K + k] = W[i];
}

__global__ void k_colstats(const float* __restrict__ X, int R, int Cc, float* __restrict__ stats) {
  int c = blockIdx.x;
  __shared__ float s1[256], s2[256];
  float a = 0.f, b = 0.f;
  for (int r = threadIdx.x; r < R; r += 256) {
    float v = X[(long)r * Cc + c];
    a += v; b += v * v;
  }
  s1[threadIdx.x] = a; s2[threadIdx.x] = b; __syncthreads();
  for (int o = 128; o > 0; o >>= 1) {
    if (threadIdx.x < o) { s1[threadIdx.x] += s1[threadIdx.x + o]; s2[threadIdx.x] += s2[threadIdx.x + o]; }
    __syncthreads();
  }
  if (threadIdx.x == 0) {
    float m = s1[0] / (float)R;
    stats[2 * c] = m;
    stats[2 * c + 1] = s2[0] / (float)R - m * m;
  }
}

__global__ void k_bn(const float* __restrict__ X, const float* __restrict__ stats,
                     const float* __restrict__ g, const float* __restrict__ b,
                     float* __restrict__ Y, long n, int Cc) {
  long i = (long)blockIdx.x * blockDim.x + threadIdx.x;
  if (i >= n) return;
  int c = (int)(i % Cc);
  float m = stats[2 * c], v = stats[2 * c + 1];
  Y[i] = (X[i] - m) * rsqrtf(v + 1e-5f) * g[c] + b[c];
}

__global__ void k_loop_acc(const float* __restrict__ ea, const int* __restrict__ ei,
                           float* __restrict__ cnt, float* __restrict__ sums, int Ee) {
  int e = blockIdx.x * blockDim.x + threadIdx.x;
  if (e >= Ee) return;
  int d = ei[Ee + e];
  atomicAdd(&cnt[d], 1.0f);
  for (int f = 0; f < 32; ++f) atomicAdd(&sums[(long)d * 32 + f], ea[(long)e * 32 + f]);
}
__global__ void k_loop_write(float* __restrict__ ea2, const float* __restrict__ cnt,
                             const float* __restrict__ sums, int Ee, int Nn) {
  int n = blockIdx.x * blockDim.x + threadIdx.x;
  if (n >= Nn) return;
  float c = cnt[n];
  float inv = (c > 0.f) ? 1.0f / fmaxf(c, 1.0f) : 0.0f;
  for (int f = 0; f < 32; ++f)
    ea2[(long)(Ee + n) * 32 + f] = (c > 0.f) ? sums[(long)n * 32 + f] * inv : 0.0f;
}

__global__ void k_wea(const float* __restrict__ We, const float* __restrict__ aedge,
                      float* __restrict__ wea) {
  int t = blockIdx.x * blockDim.x + threadIdx.x;
  if (t >= 32 * HHD) return;
  int fe = t >> 2, h = t & 3;
  float s = 0.f;
  for (int c = 0; c < CCH; ++c) s += We[(long)fe * DDM + h * CCH + c] * aedge[h * CCH + c];
  wea[fe * HHD + h] = s;
}

__global__ void k_scores(const float* __restrict__ hp, const float* __restrict__ asrc,
                         const float* __restrict__ adst, float* __restrict__ hs,
                         float* __restrict__ hd, int Nn) {
  int n = blockIdx.x * blockDim.x + threadIdx.x;
  if (n >= Nn) return;
  for (int h = 0; h < HHD; ++h) {
    float a = 0.f, b = 0.f;
    for (int c = 0; c < CCH; ++c) {
      float v = hp[(long)n * DDM + h * CCH + c];
      a += v * asrc[h * CCH + c];
      b += v * adst[h * CCH + c];
    }
    hs[n * HHD + h] = a; hd[n * HHD + h] = b;
  }
}

__global__ void k_logits(const int* __restrict__ ei, const float* __restrict__ hs,
                         const float* __restrict__ hd, float* __restrict__ lg,
                         float* __restrict__ mx, int Ee, int M2) {
  int i = blockIdx.x * blockDim.x + threadIdx.x;
  if (i >= M2) return;
  int s = (i < Ee) ? ei[i] : (i - Ee);
  int d = (i < Ee) ? ei[Ee + i] : (i - Ee);
  for (int h = 0; h < HHD; ++h) {
    float l = hs[s * HHD + h] + hd[d * HHD + h] + lg[i * HHD + h];
    l = (l > 0.f) ? l : 0.2f * l;
    lg[i * HHD + h] = l;
    atomicMaxF(&mx[d * HHD + h], l);
  }
}
__global__ void k_expw(const int* __restrict__ ei, const float* __restrict__ lg,
                       const float* __restrict__ mx, float* __restrict__ w,
                       float* __restrict__ denom, int Ee, int M2) {
  int i = blockIdx.x * blockDim.x + threadIdx.x;
  if (i >= M2) return;
  int d = (i < Ee) ? ei[Ee + i] : (i - Ee);
  for (int h = 0; h < HHD; ++h) {
    float v = __expf(lg[i * HHD + h] - mx[d * HHD + h]);
    w[i * HHD + h] = v;
    atomicAdd(&denom[d * HHD + h], v);
  }
}
__global__ void k_alphadiv(const int* __restrict__ ei, float* __restrict__ w,
                           const float* __restrict__ denom, int Ee, int M2) {
  int i = blockIdx.x * blockDim.x + threadIdx.x;
  if (i >= M2) return;
  int d = (i < Ee) ? ei[Ee + i] : (i - Ee);
  for (int h = 0; h < HHD; ++h) w[i * HHD + h] /= denom[d * HHD + h];
}

__global__ void k_agg(const float* __restrict__ hp, const float* __restrict__ alpha,
                      const int* __restrict__ ei, float* __restrict__ outp, int Ee, int M2) {
  int i = blockIdx.x * blockDim.x + threadIdx.x;
  if (i >= M2) return;
  int s = (i < Ee) ? ei[i] : (i - Ee);
  int d = (i < Ee) ? ei[Ee + i] : (i - Ee);
  for (int h = 0; h < HHD; ++h) {
    float al = alpha[i * HHD + h];
    const float* srcp = hp + (long)s * DDM + h * CCH;
    float* dstp = outp + (long)d * DDM + h * CCH;
    for (int c = 0; c < CCH; ++c) atomicAdd(&dstp[c], srcp[c] * al);
  }
}
__global__ void k_bias_elu(float* __restrict__ hbuf, const float* __restrict__ bias, long n) {
  long i = (long)blockIdx.x * blockDim.x + threadIdx.x;
  if (i >= n) return;
  float v = hbuf[i] + bias[i % DDM];
  hbuf[i] = (v > 0.f) ? v : (__expf(v) - 1.0f);
}

__global__ void k_emap(const int* __restrict__ ei, int* __restrict__ emap, int Ee, int M2, int Nn) {
  int i = blockIdx.x * blockDim.x + threadIdx.x;
  if (i >= M2) return;
  int s = (i < Ee) ? ei[i] : (i - Ee);
  int d = (i < Ee) ? ei[Ee + i] : (i - Ee);
  emap[(long)s * Nn + d] = i;
}

__global__ void k_pack(const float* __restrict__ mask, int Rr, int Ss, int LM,
                       int* __restrict__ nfs, int* __restrict__ lens) {
  int s = blockIdx.x * blockDim.x + threadIdx.x;
  if (s >= Ss) return;
  int len = 0;
  for (int n = 0; n < Rr; ++n)
    if (mask[(long)n * Ss + s] > 0.5f) { if (len < LM) nfs[(long)s * LM + len] = n; ++len; }
  lens[s] = (len > LM) ? LM : len;
}
__global__ void k_rowcnt(const float* __restrict__ m, int Ss, int* __restrict__ rc, int Nn) {
  int n = blockIdx.x * blockDim.x + threadIdx.x;
  if (n >= Nn) return;
  int c = 0;
  for (int s = 0; s < Ss; ++s) c += (m[(long)n * Ss + s] > 0.5f) ? 1 : 0;
  rc[n] = c;
}
__global__ void k_prefix(const int* __restrict__ rc, int* __restrict__ pre, int n) {
  if (blockIdx.x == 0 && threadIdx.x == 0) {
    int s = 0;
    for (int i = 0; i < n; ++i) { pre[i] = s; s += rc[i]; }
    pre[n] = s;
  }
}
__global__ void k_snkbyrank(const float* __restrict__ snk, const int* __restrict__ pre,
                            int* __restrict__ sbr, int Nn, int Ss, int cap) {
  int n = blockIdx.x * blockDim.x + threadIdx.x;
  if (n >= Nn) return;
  int r = pre[n];
  for (int s = 0; s < Ss; ++s)
    if (snk[(long)n * Ss + s] > 0.5f) { if (r < cap) sbr[r] = n; ++r; }
}
__global__ void k_pair(const float* __restrict__ srcm, const int* __restrict__ pre,
                       const int* __restrict__ sbr, int* __restrict__ paired,
                       int Nn, int Ss, int cap) {
  int n = blockIdx.x * blockDim.x + threadIdx.x;
  if (n >= Nn) return;
  int r = pre[n];
  for (int s = 0; s < Ss; ++s)
    if (srcm[(long)n * Ss + s] > 0.5f) { paired[(long)n * Ss + s] = (r < cap) ? sbr[r] : 0; ++r; }
}
__global__ void k_pack_splice(const float* __restrict__ srcm, const int* __restrict__ paired,
                              int* __restrict__ sn, int* __restrict__ tn, int* __restrict__ lens,
                              int Nn, int Ss, int LM) {
  int s = blockIdx.x * blockDim.x + threadIdx.x;
  if (s >= Ss) return;
  int len = 0;
  for (int n = 0; n < Nn; ++n)
    if (srcm[(long)n * Ss + s] > 0.5f) {
      if (len < LM) { sn[(long)s * LM + len] = n; tn[(long)s * LM + len] = paired[(long)n * Ss + s]; }
      ++len;
    }
  lens[s] = (len > LM) ? LM : len;
}
__global__ void k_att(const int* __restrict__ sn, const int* __restrict__ tn,
                      const int* __restrict__ lens, const int* __restrict__ emap,
                      const float* __restrict__ alpha, float* __restrict__ att, int Nn, int STr) {
  int r = blockIdx.x * blockDim.x + threadIdx.x;
  if (r >= STr) return;
  int s = r >> 8, t = r & (LMAXC - 1);
  if (t < lens[s]) {
    int e = emap[(long)sn[r] * Nn + tn[r]];
    for (int h = 0; h < HHD; ++h) att[r * HHD + h] = alpha[e * HHD + h];
  } else {
    for (int h = 0; h < HHD; ++h) att[r * HHD + h] = 0.f;
  }
}
__global__ void k_bsum(const float* a, const float* b, float* c) {
  int i = threadIdx.x;
  if (i < 4 * HLU) c[i] = a[i] + b[i];
}

__device__ __forceinline__ v16bf frag16(float4 x0, float4 x1, float4 y0, float4 y1) {
  v16bf a;
  a[0] = (bf16_t)x0.x;  a[1] = (bf16_t)x0.y;  a[2] = (bf16_t)x0.z;  a[3] = (bf16_t)x0.w;
  a[4] = (bf16_t)x1.x;  a[5] = (bf16_t)x1.y;  a[6] = (bf16_t)x1.z;  a[7] = (bf16_t)x1.w;
  a[8] = (bf16_t)y0.x;  a[9] = (bf16_t)y0.y;  a[10] = (bf16_t)y0.z; a[11] = (bf16_t)y0.w;
  a[12] = (bf16_t)y1.x; a[13] = (bf16_t)y1.y; a[14] = (bf16_t)y1.z; a[15] = (bf16_t)y1.w;
  return a;
}
__device__ __forceinline__ v16bf load_a_frag(const float* __restrict__ p, int khA) {
  const float4* q0 = (const float4*)(p + khA);
  const float4* q1 = (const float4*)(p + 16 + khA);
  return frag16(q0[0], q0[1], q1[0], q1[1]);
}
__device__ __forceinline__ v16bf load_b_frag(const float* __restrict__ p, int khB) {
  const float4* q0 = (const float4*)(p + (khB >> 1));
  const float4* q1 = (const float4*)(p + 16 + (khB >> 1));
  return frag16(q0[0], q0[1], q1[0], q1[1]);
}

__global__ void k_gemm_nt(const float* __restrict__ A, const float* __restrict__ BT,
                          float* __restrict__ C, int M, int Nc, int K,
                          const int* __restrict__ rowmap, const float* __restrict__ bias) {
  int lane = threadIdx.x & 31;
  int hv = lane >> 4, l16 = lane & 15;
  int khA = hv * 8, khB = hv * 16;
  int arow_idx = blockIdx.x * 16 + l16;
  int ar = (arow_idx < M) ? arow_idx : (M - 1);
  int arow = rowmap ? rowmap[ar] : ar;
  const float* Arow = A + (long)arow * K;
  int nbase = blockIdx.y * 64;
  const float* Brow0; const float* Brow1; const float* Brow2; const float* Brow3;
  {
    int b0 = nbase + l16;        if (b0 >= Nc) b0 = Nc - 1;
    int b1 = nbase + 16 + l16;   if (b1 >= Nc) b1 = Nc - 1;
    int b2 = nbase + 32 + l16;   if (b2 >= Nc) b2 = Nc - 1;
    int b3 = nbase + 48 + l16;   if (b3 >= Nc) b3 = Nc - 1;
    Brow0 = BT + (long)b0 * K; Brow1 = BT + (long)b1 * K;
    Brow2 = BT + (long)b2 * K; Brow3 = BT + (long)b3 * K;
  }
  v8f acc0 = {0.f,0.f,0.f,0.f,0.f,0.f,0.f,0.f};
  v8f acc1 = acc0, acc2 = acc0, acc3 = acc0;
  for (int k0 = 0; k0 < K; k0 += 32) {
    if (k0 + 32 < K) __builtin_prefetch(Arow + k0 + 32, 0, 1);
    v16bf a = load_a_frag(Arow + k0, khA);
    v16bf b0 = load_b_frag(Brow0 + k0, khB);
    v16bf b1 = load_b_frag(Brow1 + k0, khB);
    v16bf b2 = load_b_frag(Brow2 + k0, khB);
    v16bf b3 = load_b_frag(Brow3 + k0, khB);
    acc0 = __builtin_amdgcn_wmma_f32_16x16x32_bf16(false, a, false, b0, (short)0, acc0, false, false);
    acc1 = __builtin_amdgcn_wmma_f32_16x16x32_bf16(false, a, false, b1, (short)0, acc1, false, false);
    acc2 = __builtin_amdgcn_wmma_f32_16x16x32_bf16(false, a, false, b2, (short)0, acc2, false, false);
    acc3 = __builtin_amdgcn_wmma_f32_16x16x32_bf16(false, a, false, b3, (short)0, acc3, false, false);
  }
  v8f accs[4] = {acc0, acc1, acc2, acc3};
#pragma unroll
  for (int w = 0; w < 4; ++w) {
    int ccol = nbase + w * 16 + l16;
    if (ccol < Nc) {
      float bv = bias ? bias[ccol] : 0.f;
#pragma unroll
      for (int q = 0; q < 8; ++q) {
        int crow = blockIdx.x * 16 + hv * 8 + q;
        if (crow < M) C[(long)crow * Nc + ccol] = accs[w][q] + bv;
      }
    }
  }
}

__global__ void k_gemm_splice(const float* __restrict__ h1, const float* __restrict__ att,
                              const int* __restrict__ sn, const int* __restrict__ tn,
                              const int* __restrict__ lens, const float* __restrict__ BT,
                              const float* __restrict__ bias, float* __restrict__ C, int Mrows) {
  const int K = 2 * DDM, Nc = 4 * HLU;
  int lane = threadIdx.x & 31;
  int hv = lane >> 4, l16 = lane & 15;
  int khA = hv * 8, khB = hv * 16;
  int r = blockIdx.x * 16 + l16;
  int s = r >> 8, t = r & (LMAXC - 1);
  float vf = (t < lens[s]) ? 1.0f : 0.0f;
  int ns = (vf > 0.f) ? sn[r] : 0;
  int ntg = (vf > 0.f) ? tn[r] : 0;
  float ah[4];
  ah[0] = att[r * 4] * vf; ah[1] = att[r * 4 + 1] * vf;
  ah[2] = att[r * 4 + 2] * vf; ah[3] = att[r * 4 + 3] * vf;
  const float* Rs = h1 + (long)ns * DDM;
  const float* Rt = h1 + (long)ntg * DDM;
  int nbase = blockIdx.y * 64;
  const float* Brow0 = BT + (long)(nbase + l16) * K;
  const float* Brow1 = BT + (long)(nbase + 16 + l16) * K;
  const float* Brow2 = BT + (long)(nbase + 32 + l16) * K;
  const float* Brow3 = BT + (long)(nbase + 48 + l16) * K;
  v8f acc0 = {0.f,0.f,0.f,0.f,0.f,0.f,0.f,0.f};
  v8f acc1 = acc0, acc2 = acc0, acc3 = acc0;
  for (int k0 = 0; k0 < K; k0 += 32) {
    int ks1 = k0 + khA, ks2 = k0 + 16 + khA;
    const float* p1 = (ks1 < DDM) ? (Rs + ks1) : (Rt + (ks1 - DDM));
    const float* p2 = (ks2 < DDM) ? (Rs + ks2) : (Rt + (ks2 - DDM));
    float s1 = ah[((ks1 < DDM) ? ks1 : (ks1 - DDM)) >> 6];
    float s2 = ah[((ks2 < DDM) ? ks2 : (ks2 - DDM)) >> 6];
    float4 x0 = ((const float4*)p1)[0], x1 = ((const float4*)p1)[1];
    float4 y0 = ((const float4*)p2)[0], y1 = ((const float4*)p2)[1];
    x0.x *= s1; x0.y *= s1; x0.z *= s1; x0.w *= s1;
    x1.x *= s1; x1.y *= s1; x1.z *= s1; x1.w *= s1;
    y0.x *= s2; y0.y *= s2; y0.z *= s2; y0.w *= s2;
    y1.x *= s2; y1.y *= s2; y1.z *= s2; y1.w *= s2;
    v16bf a = frag16(x0, x1, y0, y1);
    v16bf b0 = load_b_frag(Brow0 + k0, khB);
    v16bf b1 = load_b_frag(Brow1 + k0, khB);
    v16bf b2 = load_b_frag(Brow2 + k0, khB);
    v16bf b3 = load_b_frag(Brow3 + k0, khB);
    acc0 = __builtin_amdgcn_wmma_f32_16x16x32_bf16(false, a, false, b0, (short)0, acc0, false, false);
    acc1 = __builtin_amdgcn_wmma_f32_16x16x32_bf16(false, a, false, b1, (short)0, acc1, false, false);
    acc2 = __builtin_amdgcn_wmma_f32_16x16x32_bf16(false, a, false, b2, (short)0, acc2, false, false);
    acc3 = __builtin_amdgcn_wmma_f32_16x16x32_bf16(false, a, false, b3, (short)0, acc3, false, false);
  }
  v8f accs[4] = {acc0, acc1, acc2, acc3};
#pragma unroll
  for (int w = 0; w < 4; ++w) {
    int ccol = nbase + w * 16 + l16;
    float bv = bias[ccol];
#pragma unroll
    for (int q = 0; q < 8; ++q) {
      int crow = blockIdx.x * 16 + hv * 8 + q;
      if (crow < Mrows) C[(long)crow * Nc + ccol] = accs[w][q] + bv;
    }
  }
}

__device__ __forceinline__ v8i load_a8(const unsigned char* __restrict__ p, int khA) {
  v2i t0 = *(const v2i*)(p + khA);
  v2i t1 = *(const v2i*)(p + 16 + khA);
  v2i t2 = *(const v2i*)(p + 32 + khA);
  v2i t3 = *(const v2i*)(p + 48 + khA);
  v8i a;
  a[0] = t0[0]; a[1] = t0[1]; a[2] = t1[0]; a[3] = t1[1];
  a[4] = t2[0]; a[5] = t2[1]; a[6] = t3[0]; a[7] = t3[1];
  return a;
}
__device__ __forceinline__ v8i load_b8(const unsigned char* __restrict__ p, int khB) {
  v2i t0 = *(const v2i*)(p + (khB >> 1));
  v2i t1 = *(const v2i*)(p + 16 + (khB >> 1));
  v2i t2 = *(const v2i*)(p + 32 + (khB >> 1));
  v2i t3 = *(const v2i*)(p + 48 + (khB >> 1));
  v8i b;
  b[0] = t0[0]; b[1] = t0[1]; b[2] = t1[0]; b[3] = t1[1];
  b[4] = t2[0]; b[5] = t2[1]; b[6] = t3[0]; b[7] = t3[1];
  return b;
}
__global__ void k_gemm_iu8(const unsigned char* __restrict__ A, const unsigned char* __restrict__ BT,
                           int* __restrict__ C, int M, int Nc, int K) {
  int lane = threadIdx.x & 31;
  int hv = lane >> 4, l16 = lane & 15;
  int khA = hv * 8, khB = hv * 16;
  const unsigned char* Ar = A + (long)(blockIdx.x * 16 + l16) * K;
  int nbase = blockIdx.y * 64;
  const unsigned char* Br0 = BT + (long)(nbase + l16) * K;
  const unsigned char* Br1 = BT + (long)(nbase + 16 + l16) * K;
  const unsigned char* Br2 = BT + (long)(nbase + 32 + l16) * K;
  const unsigned char* Br3 = BT + (long)(nbase + 48 + l16) * K;
  v8i acc0 = {0,0,0,0,0,0,0,0};
  v8i acc1 = acc0, acc2 = acc0, acc3 = acc0;
  for (int k0 = 0; k0 < K; k0 += 64) {
    if (k0 + 64 < K) __builtin_prefetch(Ar + k0 + 64, 0, 1);
    v8i a = load_a8(Ar + k0, khA);
    v8i b0 = load_b8(Br0 + k0, khB);
    v8i b1 = load_b8(Br1 + k0, khB);
    v8i b2 = load_b8(Br2 + k0, khB);
    v8i b3 = load_b8(Br3 + k0, khB);
    acc0 = __builtin_amdgcn_wmma_i32_16x16x64_iu8(false, a, false, b0, acc0, false, false);
    acc1 = __builtin_amdgcn_wmma_i32_16x16x64_iu8(false, a, false, b1, acc1, false, false);
    acc2 = __builtin_amdgcn_wmma_i32_16x16x64_iu8(false, a, false, b2, acc2, false, false);
    acc3 = __builtin_amdgcn_wmma_i32_16x16x64_iu8(false, a, false, b3, acc3, false, false);
  }
  v8i accs[4] = {acc0, acc1, acc2, acc3};
#pragma unroll
  for (int w = 0; w < 4; ++w) {
    int ccol = nbase + w * 16 + l16;
#pragma unroll
    for (int q = 0; q < 8; ++q) {
      int crow = blockIdx.x * 16 + hv * 8 + q;
      C[(long)crow * Nc + ccol] = accs[w][q];
    }
  }
}

__global__ void k_lstm(const float* __restrict__ gates, const float* __restrict__ Whh,
                       const int* __restrict__ lens, float* __restrict__ outp, int T) {
  int b = blockIdx.x, u = threadIdx.x;
  __shared__ float hs[HLU];
  float c = 0.f;
  hs[u] = 0.f;
  __syncthreads();
  int L = lens[b];
  const float* g = gates + (long)b * T * (4 * HLU);
  for (int t = 0; t < L; ++t) {
    const float* gt = g + (long)t * (4 * HLU);
    float zi = gt[u], zf = gt[HLU + u], zg = gt[2 * HLU + u], zo = gt[3 * HLU + u];
    for (int j = 0; j < HLU; ++j) {
      float hj = hs[j];
      zi += hj * Whh[(long)u * HLU + j];
      zf += hj * Whh[(long)(HLU + u) * HLU + j];
      zg += hj * Whh[(long)(2 * HLU + u) * HLU + j];
      zo += hj * Whh[(long)(3 * HLU + u) * HLU + j];
    }
    float i_ = 1.f / (1.f + __expf(-zi));
    float f_ = 1.f / (1.f + __expf(-zf));
    float o_ = 1.f / (1.f + __expf(-zo));
    float gg = tanhf(zg);
    c = f_ * c + i_ * gg;
    float hnew = o_ * tanhf(c);
    __syncthreads();
    hs[u] = hnew;
    __syncthreads();
  }
  outp[(long)b * HLU + u] = hs[u];
}

__global__ void k_fragscan(const float* __restrict__ frags, unsigned char* __restrict__ fT,
                           unsigned char* __restrict__ mT, int* __restrict__ fcnt, int Nn, int Ff) {
  int f = blockIdx.x * blockDim.x + threadIdx.x;
  if (f >= Ff) return;
  int fs = -1, fe = -1, cnt = 0;
  for (int n = 0; n < Nn; ++n) {
    bool b = frags[(long)n * Ff + f] > 0.5f;
    fT[(long)f * Nn + n] = b ? 1 : 0;
    if (b) { if (fs < 0) fs = n; fe = n; ++cnt; }
  }
  if (fs < 0) { fs = 0; fe = Nn - 1; }
  fcnt[f] = cnt;
  for (int n = 0; n < Nn; ++n) mT[(long)f * Nn + n] = (n >= fs && n <= fe) ? 1 : 0;
}
__global__ void k_pathsT(const float* __restrict__ cp, unsigned char* __restrict__ pT, int Nn, int Pp) {
  int p = blockIdx.x * blockDim.x + threadIdx.x;
  if (p >= Pp) return;
  for (int n = 0; n < Nn; ++n) pT[(long)p * Nn + n] = (cp[(long)n * Pp + p] > 0.5f) ? 1 : 0;
}
__global__ void k_posneg(const int* __restrict__ Amat, const int* __restrict__ Bmat,
                         const int* __restrict__ fcnt, const float* __restrict__ femb,
                         float* __restrict__ pos_o, float* __restrict__ neg_o, int Ff, int Pp) {
  int p = blockIdx.x, u = threadIdx.x;
  float ap = 0.f, an = 0.f;
  for (int f = 0; f < Ff; ++f) {
    int a = Amat[(long)f * Pp + p];
    if (a > 0) {
      int mism = fcnt[f] - 2 * a + Bmat[(long)f * Pp + p];
      float v = femb[(long)f * HLU + u];
      if (mism == 0) ap += v; else an += v;
    }
  }
  pos_o[(long)p * HLU + u] = ap;
  neg_o[(long)p * HLU + u] = an;
}

__global__ void k_head(const float* __restrict__ ne, const float* __restrict__ se,
                       const float* __restrict__ ab, const float* __restrict__ po,
                       const float* __restrict__ ng, const float* __restrict__ fc1W,
                       const float* __restrict__ fc1b, const float* __restrict__ fc2W,
                       const float* __restrict__ fc2b, float* __restrict__ outp) {
  int p = blockIdx.x, u = threadIdx.x;
  __shared__ float comb[4 * HLU + 1];
  __shared__ float r1[HLU];
  comb[u] = ne[(long)p * HLU + u];
  comb[HLU + u] = se[(long)p * HLU + u];
  if (u == 0) comb[2 * HLU] = ab[p];
  comb[2 * HLU + 1 + u] = po[(long)p * HLU + u];
  comb[3 * HLU + 1 + u] = ng[(long)p * HLU + u];
  __syncthreads();
  float acc = fc1b[u];
  for (int j = 0; j < 4 * HLU + 1; ++j) acc += comb[j] * fc1W[(long)u * (4 * HLU + 1) + j];
  r1[u] = (acc > 0.f) ? acc : 0.f;
  __syncthreads();
  if (u == 0) {
    float s = fc2b[0];
    for (int j = 0; j < HLU; ++j) s += r1[j] * fc2W[j];
    outp[p] = 1.f / (1.f + __expf(-s));
  }
}

extern "C" void kernel_launch(void* const* d_in, const int* in_sizes, int n_in,
                              void* d_out, int out_size, void* d_ws, size_t ws_size,
                              hipStream_t stream) {
  const float* x    = (const float*)d_in[0];
  const int*   ei   = (const int*)d_in[1];
  const float* eat  = (const float*)d_in[2];
  const float* cp   = (const float*)d_in[3];
  const float* cpsm = (const float*)d_in[4];
  const float* cpkm = (const float*)d_in[5];
  const float* ab   = (const float*)d_in[6];
  const float* fr   = (const float*)d_in[7];
  const float* bnxg = (const float*)d_in[8],  *bnxb = (const float*)d_in[9];
  const float* bneg = (const float*)d_in[10], *bneb = (const float*)d_in[11];
  const float* g0W  = (const float*)d_in[12], *g0We = (const float*)d_in[13];
  const float* g0as = (const float*)d_in[14], *g0ad = (const float*)d_in[15];
  const float* g0ae = (const float*)d_in[16], *g0b  = (const float*)d_in[17];
  const float* g1W  = (const float*)d_in[18], *g1We = (const float*)d_in[19];
  const float* g1as = (const float*)d_in[20], *g1ad = (const float*)d_in[21];
  const float* g1ae = (const float*)d_in[22], *g1b  = (const float*)d_in[23];
  const float* nWih = (const float*)d_in[24], *nWhh = (const float*)d_in[25];
  const float* nbih = (const float*)d_in[26], *nbhh = (const float*)d_in[27];
  const float* sWih = (const float*)d_in[28], *sWhh = (const float*)d_in[29];
  const float* sbih = (const float*)d_in[30], *sbhh = (const float*)d_in[31];
  const float* fWih = (const float*)d_in[32], *fWhh = (const float*)d_in[33];
  const float* fbih = (const float*)d_in[34], *fbhh = (const float*)d_in[35];
  const float* fc1W = (const float*)d_in[36], *fc1b = (const float*)d_in[37];
  const float* fc2W = (const float*)d_in[38], *fc2b = (const float*)d_in[39];
  float* outp = (float*)d_out;

  const int Nn = in_sizes[0] / 32;
  const int Ee = in_sizes[1] / 2;
  const int Pp = in_sizes[6];
  const int Ff = in_sizes[7] / Nn;
  const int M2 = Ee + Nn;
  const int STr = Pp * LMAXC;
  const int FTr = Ff * LFMAXC;

  char* wp = (char*)d_ws;
  auto alloc = [&](size_t bytes) -> void* {
    void* r = (void*)wp;
    wp += (bytes + 255) & ~(size_t)255;
    return r;
  };
  auto cdiv = [](long a, long b) -> unsigned { return (unsigned)((a + b - 1) / b); };

  float* xn     = (float*)alloc((size_t)Nn * 32 * 4);
  float* ea2    = (float*)alloc((size_t)M2 * 32 * 4);
  float* statsx = (float*)alloc(2 * 64 * 4);
  float* statse = (float*)alloc(2 * 64 * 4);
  float* cnt    = (float*)alloc((size_t)Nn * 4);
  float* sums   = (float*)alloc((size_t)Nn * 32 * 4);
  float* hproj  = (float*)alloc((size_t)Nn * DDM * 4);
  float* h0     = (float*)alloc((size_t)Nn * DDM * 4);
  float* h1     = (float*)alloc((size_t)Nn * DDM * 4);
  float* hsb    = (float*)alloc((size_t)Nn * HHD * 4);
  float* hdb    = (float*)alloc((size_t)Nn * HHD * 4);
  float* wea    = (float*)alloc(32 * HHD * 4);
  float* weaT   = (float*)alloc(32 * HHD * 4);
  float* WT     = (float*)alloc((size_t)DDM * DDM * 4);
  float* lg     = (float*)alloc((size_t)M2 * HHD * 4);
  float* mx     = (float*)alloc((size_t)Nn * HHD * 4);
  float* denom  = (float*)alloc((size_t)Nn * HHD * 4);
  float* alpha0 = (float*)alloc((size_t)M2 * HHD * 4);
  float* alpha1 = (float*)alloc((size_t)M2 * HHD * 4);
  int*   emap   = (int*)alloc((size_t)Nn * Nn * 4);
  int*   nfsN   = (int*)alloc((size_t)STr * 4);
  int*   lenN   = (int*)alloc((size_t)Pp * 4);
  int*   rc     = (int*)alloc((size_t)Nn * 4);
  int*   preS   = (int*)alloc((size_t)(Nn + 1) * 4);
  int*   preK   = (int*)alloc((size_t)(Nn + 1) * 4);
  int*   sbr    = (int*)alloc((size_t)(STr + 1) * 4);
  int*   paired = (int*)alloc((size_t)Nn * Pp * 4);
  int*   snb    = (int*)alloc((size_t)STr * 4);
  int*   tnb    = (int*)alloc((size_t)STr * 4);
  int*   lenS   = (int*)alloc((size_t)Pp * 4);
  float* att    = (float*)alloc((size_t)STr * HHD * 4);
  int*   nfsF   = (int*)alloc((size_t)FTr * 4);
  int*   lenF   = (int*)alloc((size_t)Ff * 4);
  float* bsum   = (float*)alloc(4 * HLU * 4);
  float* gates  = (float*)alloc((size_t)STr * (4 * HLU) * 4);
  float* nemb   = (float*)alloc((size_t)Pp * HLU * 4);
  float* semb   = (float*)alloc((size_t)Pp * HLU * 4);
  float* femb   = (float*)alloc((size_t)Ff * HLU * 4);
  unsigned char* pT = (unsigned char*)alloc((size_t)Pp * Nn);
  unsigned char* fT = (unsigned char*)alloc((size_t)Ff * Nn);
  unsigned char* mT = (unsigned char*)alloc((size_t)Ff * Nn);
  int*   fcnt   = (int*)alloc((size_t)Ff * 4);
  int*   Amat   = (int*)alloc((size_t)Ff * Pp * 4);
  int*   Bmat   = (int*)alloc((size_t)Ff * Pp * 4);
  float* pos_o  = (float*)alloc((size_t)Pp * HLU * 4);
  float* neg_o  = (float*)alloc((size_t)Pp * HLU * 4);

  k_colstats<<<32, 256, 0, stream>>>(x, Nn, 32, statsx);
  k_bn<<<cdiv((long)Nn * 32, 256), 256, 0, stream>>>(x, statsx, bnxg, bnxb, xn, (long)Nn * 32, 32);
  k_colstats<<<32, 256, 0, stream>>>(eat, Ee, 32, statse);
  k_bn<<<cdiv((long)Ee * 32, 256), 256, 0, stream>>>(eat, statse, bneg, bneb, ea2, (long)Ee * 32, 32);

  k_fillf<<<cdiv(Nn, 256), 256, 0, stream>>>(cnt, 0.f, Nn);
  k_fillf<<<cdiv((long)Nn * 32, 256), 256, 0, stream>>>(sums, 0.f, (long)Nn * 32);
  k_loop_acc<<<cdiv(Ee, 256), 256, 0, stream>>>(ea2, ei, cnt, sums, Ee);
  k_loop_write<<<cdiv(Nn, 256), 256, 0, stream>>>(ea2, cnt, sums, Ee, Nn);

  const float* layW[2]  = {g0W, g1W};
  const float* layWe[2] = {g0We, g1We};
  const float* layAs[2] = {g0as, g1as};
  const float* layAd[2] = {g0ad, g1ad};
  const float* layAe[2] = {g0ae, g1ae};
  const float* layB[2]  = {g0b, g1b};
  float* layIn[2]  = {xn, h0};
  float* layOut[2] = {h0, h1};
  float* layAl[2]  = {alpha0, alpha1};
  int layK[2] = {32, DDM};
  for (int L = 0; L < 2; ++L) {
    k_transpose<<<cdiv((long)layK[L] * DDM, 256), 256, 0, stream>>>(layW[L], WT, layK[L], DDM);
    dim3 gP(cdiv(Nn, 16), cdiv(DDM, 64));
    k_gemm_nt<<<gP, 32, 0, stream>>>(layIn[L], WT, hproj, Nn, DDM, layK[L], nullptr, nullptr);
    k_wea<<<1, 128, 0, stream>>>(layWe[L], layAe[L], wea);
    k_transpose<<<1, 128, 0, stream>>>(wea, weaT, 32, HHD);
    dim3 gE(cdiv(M2, 16), 1);
    k_gemm_nt<<<gE, 32, 0, stream>>>(ea2, weaT, lg, M2, HHD, 32, nullptr, nullptr);
    k_scores<<<cdiv(Nn, 256), 256, 0, stream>>>(hproj, layAs[L], layAd[L], hsb, hdb, Nn);
    k_fillf<<<cdiv((long)Nn * HHD, 256), 256, 0, stream>>>(mx, -1e30f, (long)Nn * HHD);
    k_logits<<<cdiv(M2, 256), 256, 0, stream>>>(ei, hsb, hdb, lg, mx, Ee, M2);
    k_fillf<<<cdiv((long)Nn * HHD, 256), 256, 0, stream>>>(denom, 0.f, (long)Nn * HHD);
    k_expw<<<cdiv(M2, 256), 256, 0, stream>>>(ei, lg, mx, layAl[L], denom, Ee, M2);
    k_alphadiv<<<cdiv(M2, 256), 256, 0, stream>>>(ei, layAl[L], denom, Ee, M2);
    k_fillf<<<cdiv((long)Nn * DDM, 256), 256, 0, stream>>>(layOut[L], 0.f, (long)Nn * DDM);
    k_agg<<<cdiv(M2, 256), 256, 0, stream>>>(hproj, layAl[L], ei, layOut[L], Ee, M2);
    k_bias_elu<<<cdiv((long)Nn * DDM, 256), 256, 0, stream>>>(layOut[L], layB[L], (long)Nn * DDM);
  }

  k_filli<<<cdiv((long)Nn * Nn, 256), 256, 0, stream>>>(emap, 0, (long)Nn * Nn);
  k_emap<<<cdiv(M2, 256), 256, 0, stream>>>(ei, emap, Ee, M2, Nn);

  k_pack<<<cdiv(Pp, 256), 256, 0, stream>>>(cp, Nn, Pp, LMAXC, nfsN, lenN);
  k_rowcnt<<<cdiv(Nn, 256), 256, 0, stream>>>(cpsm, Pp, rc, Nn);
  k_prefix<<<1, 1, 0, stream>>>(rc, preS, Nn);
  k_rowcnt<<<cdiv(Nn, 256), 256, 0, stream>>>(cpkm, Pp, rc, Nn);
  k_prefix<<<1, 1, 0, stream>>>(rc, preK, Nn);
  k_filli<<<cdiv(STr + 1, 256), 256, 0, stream>>>(sbr, 0, STr + 1);
  k_snkbyrank<<<cdiv(Nn, 256), 256, 0, stream>>>(cpkm, preK, sbr, Nn, Pp, STr + 1);
  k_pair<<<cdiv(Nn, 256), 256, 0, stream>>>(cpsm, preS, sbr, paired, Nn, Pp, STr + 1);
  k_pack_splice<<<cdiv(Pp, 256), 256, 0, stream>>>(cpsm, paired, snb, tnb, lenS, Nn, Pp, LMAXC);
  k_att<<<cdiv(STr, 256), 256, 0, stream>>>(snb, tnb, lenS, emap, alpha1, att, Nn, STr);
  k_pack<<<cdiv(Ff, 256), 256, 0, stream>>>(fr, Nn, Ff, LFMAXC, nfsF, lenF);

  k_bsum<<<1, 256, 0, stream>>>(nbih, nbhh, bsum);
  {
    dim3 g(cdiv(STr, 16), cdiv(4 * HLU, 64));
    k_gemm_nt<<<g, 32, 0, stream>>>(h1, nWih, gates, STr, 4 * HLU, DDM, nfsN, bsum);
  }
  k_lstm<<<Pp, HLU, 0, stream>>>(gates, nWhh, lenN, nemb, LMAXC);

  k_bsum<<<1, 256, 0, stream>>>(sbih, sbhh, bsum);
  {
    dim3 g(cdiv(STr, 16), cdiv(4 * HLU, 64));
    k_gemm_splice<<<g, 32, 0, stream>>>(h1, att, snb, tnb, lenS, sWih, bsum, gates, STr);
  }
  k_lstm<<<Pp, HLU, 0, stream>>>(gates, sWhh, lenS, semb, LMAXC);

  k_bsum<<<1, 256, 0, stream>>>(fbih, fbhh, bsum);
  {
    dim3 g(cdiv(FTr, 16), cdiv(4 * HLU, 64));
    k_gemm_nt<<<g, 32, 0, stream>>>(h1, fWih, gates, FTr, 4 * HLU, DDM, nfsF, bsum);
  }
  k_lstm<<<Ff, HLU, 0, stream>>>(gates, fWhh, lenF, femb, LFMAXC);

  k_fragscan<<<cdiv(Ff, 256), 256, 0, stream>>>(fr, fT, mT, fcnt, Nn, Ff);
  k_pathsT<<<cdiv(Pp, 256), 256, 0, stream>>>(cp, pT, Nn, Pp);
  {
    dim3 g(cdiv(Ff, 16), cdiv(Pp, 64));
    k_gemm_iu8<<<g, 32, 0, stream>>>(fT, pT, Amat, Ff, Pp, Nn);
    k_gemm_iu8<<<g, 32, 0, stream>>>(mT, pT, Bmat, Ff, Pp, Nn);
  }
  k_posneg<<<Pp, HLU, 0, stream>>>(Amat, Bmat, fcnt, femb, pos_o, neg_o, Ff, Pp);

  k_head<<<Pp, HLU, 0, stream>>>(nemb, semb, ab, pos_o, neg_o, fc1W, fc1b, fc2W, fc2b, outp);
}
